// NoisyTopkRouter_51874615001519
// MI455X (gfx1250) — matrix operand order repaired and checked by reading, not compiled, not run on hardware
//
#include <hip/hip_runtime.h>
#include <hip/hip_bf16.h>
#include <math.h>

#define TOPK 8
#define N_EXP 64
#define N_TOK 16384
#define N_EMB 4096

#define ROWS_PER_BLOCK 128   // 8 waves x 16 rows
#define KTILE 32
#define NKT   (N_EMB / KTILE)
#define A_STRIDE 40          // bf16 elems per LDS row (32 + 8 pad) -> 80B stride
#define B_STRIDE 40
#define NL_STRIDE 68         // f32 elems per noisy-logit LDS row (64 + 4 pad)

// ws layout (bytes):
//   [0   .. 515]            : probs_sum[64] f32, freq[64] f32, lse2 f32
//   [1024 .. 1024+512K)     : w_route bf16, layout [kt][c][k_local] (kt*2048 + c*32 + k)
//   [1024+512K .. 1024+1M)  : w_noise bf16, same layout
#define WS_WB_OFF   1024
#define WS_WB_BYTES (N_EMB * N_EXP * 2)

typedef __attribute__((ext_vector_type(16))) __bf16 v16bf;
typedef __attribute__((ext_vector_type(2)))  __bf16 v2bf;
typedef __attribute__((ext_vector_type(8)))  float  v8f;
typedef __attribute__((ext_vector_type(4)))  int    v4i;

union FragU { uint4 u[2]; v16bf v; };

__device__ inline unsigned pk_bf16(float a, float b) {
#if __has_builtin(__builtin_amdgcn_cvt_pk_bf16_f32)
    union { v2bf v; unsigned u; } t;
    t.v = __builtin_amdgcn_cvt_pk_bf16_f32(a, b);   // v_cvt_pk_bf16_f32
    return t.u;
#else
    union { __bf16 h[2]; unsigned u; } t;
    t.h[0] = (__bf16)a; t.h[1] = (__bf16)b;
    return t.u;
#endif
}

#if __has_builtin(__builtin_amdgcn_global_load_async_to_lds_b128)
#define HAVE_ASYNC_LDS 1
#else
#define HAVE_ASYNC_LDS 0
#endif

__device__ inline void async_copy16(const void* g, void* l) {
#if HAVE_ASYNC_LDS
    __builtin_amdgcn_global_load_async_to_lds_b128((v4i*)const_cast<void*>(g), (v4i*)l, 0, 0);
#else
    *(uint4*)l = *(const uint4*)g;        // global_load_b128 + ds_store_b128
#endif
}

__device__ inline void async_wait_all() {
#if HAVE_ASYNC_LDS
#if __has_builtin(__builtin_amdgcn_s_wait_asynccnt)
    __builtin_amdgcn_s_wait_asynccnt(0);
#else
    asm volatile("s_wait_asynccnt 0" ::: "memory");
#endif
#endif
}

// ---- one-time weight convert: f32 [k][c] -> bf16 [kt][c][k_local] ----
__global__ __launch_bounds__(256)
void convert_weights(const float* __restrict__ w_route, const float* __restrict__ w_noise,
                     unsigned short* __restrict__ wbR, unsigned short* __restrict__ wbN)
{
    const int i = blockIdx.x * 256 + threadIdx.x;   // 0 .. 262143, coalesced reads
    const int k = i >> 6;
    const int c = i & 63;
    const int o = (k >> 5) * (N_EXP * KTILE) + c * KTILE + (k & 31);
    union { __bf16 h; unsigned short u; } cv;
    cv.h = (__bf16)w_route[i]; wbR[o] = cv.u;
    cv.h = (__bf16)w_noise[i]; wbN[o] = cv.u;
}

__global__ __launch_bounds__(256)
void router_main(const float* __restrict__ x,
                 const float* __restrict__ b_route, const float* __restrict__ b_noise,
                 const float* __restrict__ noise,
                 const unsigned short* __restrict__ wbR,
                 const unsigned short* __restrict__ wbN,
                 float* __restrict__ out_router, float* __restrict__ out_idx,
                 float* __restrict__ ws_probs, float* __restrict__ ws_freq,
                 float* __restrict__ ws_lse2)
{
    __shared__ __align__(16) __bf16 sA[2][ROWS_PER_BLOCK * A_STRIDE];   // 2 x 10240 B
    __shared__ __align__(16) __bf16 sBr[2][N_EXP * B_STRIDE];           // 2 x  5120 B
    __shared__ __align__(16) __bf16 sBn[2][N_EXP * B_STRIDE];           // 2 x  5120 B
    __shared__ __align__(16) float  sNL[ROWS_PER_BLOCK * NL_STRIDE];    // 34816 B
    __shared__ float  sProbs[N_EXP];
    __shared__ float  sFreq[N_EXP];
    __shared__ float  sLse2;

    const int tid   = threadIdx.x;
    const int wave  = tid >> 5;
    const int lane  = tid & 31;
    const int lan15 = lane & 15;
    const int selhi = lane >> 4;                  // 0 for lanes 0-15, 1 for 16-31
    const int row0  = blockIdx.x * ROWS_PER_BLOCK;

    if (tid < N_EXP) { sProbs[tid] = 0.f; sFreq[tid] = 0.f; }
    if (tid == 0)    { sLse2 = 0.f; }

    const v8f vzero = {0.f,0.f,0.f,0.f,0.f,0.f,0.f,0.f};
    v8f accR[4], accN[4];
#pragma unroll
    for (int t = 0; t < 4; ++t) { accR[t] = vzero; accN[t] = vzero; }

    // ---- per-thread staging maps (constant over the K loop) ----
    const float* xPtr[4];
    int aOff[4];                                  // LDS element offset within a buffer
#pragma unroll
    for (int i = 0; i < 4; ++i) {
        const int e = (i * 256 + tid) << 2;       // element in 128x32 tile
        const int r = e >> 5;
        const int k = e & 31;
        xPtr[i]  = x + (size_t)(row0 + r) * N_EMB + k;
        aOff[i]  = r * A_STRIDE + k;
    }
    const int bc  = tid >> 2;                     // B column 0..63
    const int bkc = tid & 3;                      // 16B chunk within 64B k-run
    const int bOffL = bc * B_STRIDE + bkc * 8;    // LDS elem offset
    const size_t bOffG = (size_t)bc * KTILE + bkc * 8;

    float4 aReg[4];

    // ---- prologue: stage tile 0 into buffer 0 ----
#pragma unroll
    for (int i = 0; i < 4; ++i) aReg[i] = *(const float4*)(xPtr[i]);
    async_copy16(wbR + bOffG, &sBr[0][bOffL]);
    async_copy16(wbN + bOffG, &sBn[0][bOffL]);
#pragma unroll
    for (int i = 0; i < 4; ++i) {
        uint2 p;
        p.x = pk_bf16(aReg[i].x, aReg[i].y);
        p.y = pk_bf16(aReg[i].z, aReg[i].w);
        *(uint2*)&sA[0][aOff[i]] = p;
    }
    async_wait_all();

    for (int kt = 0; kt < NKT; ++kt) {
        const int cur  = kt & 1;
        const int nxt  = cur ^ 1;
        const bool more = (kt + 1) < NKT;
        __syncthreads();                          // buffer 'cur' staged for everyone

        // ---- kick off next tile: A global loads + async B into 'nxt' ----
        if (more) {
            const size_t gk = (size_t)(kt + 1) * KTILE;
#pragma unroll
            for (int i = 0; i < 4; ++i) aReg[i] = *(const float4*)(xPtr[i] + gk);
            const size_t wOff = (size_t)(kt + 1) * (N_EXP * KTILE) + bOffG;
            async_copy16(wbR + wOff, &sBr[nxt][bOffL]);
            async_copy16(wbN + wOff, &sBn[nxt][bOffL]);
        }

        // ---- compute on 'cur' ----
        FragU af;
        const __bf16* arow = &sA[cur][(wave * 16 + lan15) * A_STRIDE + selhi * 16];
        af.u[0] = *(const uint4*)(arow);
        af.u[1] = *(const uint4*)(arow + 8);

#pragma unroll
        for (int t = 0; t < 4; ++t) {
            FragU bfr;
            const __bf16* bcol = &sBr[cur][(t * 16 + lan15) * B_STRIDE + selhi * 16];
            bfr.u[0] = *(const uint4*)(bcol);
            bfr.u[1] = *(const uint4*)(bcol + 8);
            accR[t] = __builtin_amdgcn_wmma_f32_16x16x32_bf16(
                false, af.v, false, bfr.v, (short)0, accR[t], false, false);

            const __bf16* ncol = &sBn[cur][(t * 16 + lan15) * B_STRIDE + selhi * 16];
            bfr.u[0] = *(const uint4*)(ncol);
            bfr.u[1] = *(const uint4*)(ncol + 8);
            accN[t] = __builtin_amdgcn_wmma_f32_16x16x32_bf16(
                false, af.v, false, bfr.v, (short)0, accN[t], false, false);
        }

        // ---- convert + store next A tile into 'nxt' ----
        if (more) {
#pragma unroll
            for (int i = 0; i < 4; ++i) {
                uint2 p;
                p.x = pk_bf16(aReg[i].x, aReg[i].y);
                p.y = pk_bf16(aReg[i].z, aReg[i].w);
                *(uint2*)&sA[nxt][aOff[i]] = p;
            }
        }
        async_wait_all();
    }

    // ---- epilogue part 1: bias + softplus noise mix -> noisy logits in LDS ----
#pragma unroll
    for (int t = 0; t < 4; ++t) {
        const int col = t * 16 + lan15;
        const float br = b_route[col];
        const float bn = b_noise[col];
#pragma unroll
        for (int r = 0; r < 8; ++r) {
            const int rl  = wave * 16 + selhi * 8 + r;  // C/D layout: lane-half picks M+8
            const int row = row0 + rl;
            const float lg = accR[t][r] + br;
            const float nl = accN[t][r] + bn;
            const float sp = (nl > 15.f) ? nl : __logf(1.f + __expf(nl));
            const float nz = noise[(size_t)row * N_EXP + col];
            sNL[rl * NL_STRIDE + col] = lg + nz * sp;
        }
    }
    __syncthreads();

    // ---- epilogue part 2: per-row top-8 / softmaxes (one lane per row) ----
    if (lane < 16) {
        const int rl  = wave * 16 + lane;
        const int row = row0 + rl;
        float* nr = &sNL[rl * NL_STRIDE];

        float mx = nr[0];
        for (int e = 1; e < N_EXP; ++e) mx = fmaxf(mx, nr[e]);
        float s = 0.f;
        for (int e = 0; e < N_EXP; ++e) s += __expf(nr[e] - mx);
        const float inv = 1.f / s;
        for (int e = 0; e < N_EXP; ++e) atomicAdd(&sProbs[e], __expf(nr[e] - mx) * inv);
        const float lse = mx + __logf(s);

        float tv[TOPK]; int ti[TOPK];
#pragma unroll
        for (int j = 0; j < TOPK; ++j) {
            float best = -1e30f; int bi = 0;
            for (int e = 0; e < N_EXP; ++e) {
                const float v = nr[e];
                if (v > best) { best = v; bi = e; }
            }
            tv[j] = best; ti[j] = bi; nr[bi] = -1e30f;
        }
        float ss = 0.f;
#pragma unroll
        for (int j = 0; j < TOPK; ++j) ss += __expf(tv[j] - tv[0]);
        const float sinv = 1.f / ss;

        for (int e = 0; e < N_EXP; ++e) nr[e] = 0.f;
#pragma unroll
        for (int j = 0; j < TOPK; ++j) {
            nr[ti[j]] = __expf(tv[j] - tv[0]) * sinv;
            atomicAdd(&sFreq[ti[j]], 1.f);
            out_idx[(size_t)row * TOPK + j] = (float)ti[j];
        }
        atomicAdd(&sLse2, lse * lse);
    }
    __syncthreads();

    // ---- coalesced writeback of router_output (8 float4 per thread) ----
#pragma unroll
    for (int i = 0; i < 8; ++i) {
        const int v4 = i * 256 + tid;            // 0..2047
        const int e  = v4 << 2;
        const int rl = e >> 6;
        const int c  = e & 63;
        const float* src = &sNL[rl * NL_STRIDE + c];
        float4 o; o.x = src[0]; o.y = src[1]; o.z = src[2]; o.w = src[3];
        *(float4*)(out_router + (size_t)(row0 + rl) * N_EXP + c) = o;
    }

    // ---- reduce per-block partials to global workspace ----
    if (tid < N_EXP) {
        atomicAdd(&ws_probs[tid], sProbs[tid]);
        atomicAdd(&ws_freq[tid],  sFreq[tid]);
    }
    if (tid == 0) atomicAdd(ws_lse2, sLse2);
}

__global__ __launch_bounds__(64)
void router_aux(const float* __restrict__ ws_probs, const float* __restrict__ ws_freq,
                const float* __restrict__ ws_lse2, float* __restrict__ out_aux)
{
    __shared__ float sp[N_EXP], sf[N_EXP];
    const int t = threadIdx.x;
    sp[t] = ws_probs[t];
    sf[t] = ws_freq[t];
    __syncthreads();
    if (t == 0) {
        float s1 = 0.f, s2 = 0.f;
        for (int e = 0; e < N_EXP; ++e) { s1 += fabsf(sp[e]); s2 += fabsf(sf[e]); }
        s1 = fmaxf(s1, 1e-12f); s2 = fmaxf(s2, 1e-12f);
        float dot = 0.f;
        for (int e = 0; e < N_EXP; ++e) dot += (sp[e] / s1) * (sf[e] / s2);
        const float switchloss = (float)N_EXP * dot;
        const float zloss = ws_lse2[0] / (float)N_TOK;
        out_aux[0] = switchloss + 0.1f * zloss;
    }
}

extern "C" void kernel_launch(void* const* d_in, const int* in_sizes, int n_in,
                              void* d_out, int out_size, void* d_ws, size_t ws_size,
                              hipStream_t stream)
{
    const float* x       = (const float*)d_in[0];
    const float* w_route = (const float*)d_in[1];
    const float* b_route = (const float*)d_in[2];
    const float* w_noise = (const float*)d_in[3];
    const float* b_noise = (const float*)d_in[4];
    const float* noise   = (const float*)d_in[5];

    float* out        = (float*)d_out;
    float* out_router = out;                                    // 16384*64
    float* out_idx    = out + (size_t)N_TOK * N_EXP;            // 16384*8
    float* out_aux    = out_idx + (size_t)N_TOK * TOPK;         // 1

    float* ws       = (float*)d_ws;
    float* ws_probs = ws;
    float* ws_freq  = ws + N_EXP;
    float* ws_lse2  = ws + 2 * N_EXP;
    unsigned short* wbR = (unsigned short*)((char*)d_ws + WS_WB_OFF);
    unsigned short* wbN = (unsigned short*)((char*)d_ws + WS_WB_OFF + WS_WB_BYTES);

    (void)hipMemsetAsync(d_ws, 0, (2 * N_EXP + 1) * sizeof(float), stream);

    convert_weights<<<(N_EMB * N_EXP) / 256, 256, 0, stream>>>(w_route, w_noise, wbR, wbN);

    router_main<<<N_TOK / ROWS_PER_BLOCK, 256, 0, stream>>>(
        x, b_route, b_noise, noise, wbR, wbN,
        out_router, out_idx, ws_probs, ws_freq, ws_lse2);

    router_aux<<<1, 64, 0, stream>>>(ws_probs, ws_freq, ws_lse2, out_aux);
}
